// IDSSModule_2327872275003
// MI455X (gfx1250) — matrix operand order repaired and checked by reading, not compiled, not run on hardware
//
#include <hip/hip_runtime.h>
#include <hip/hip_bf16.h>

// ---------------------------------------------------------------------------
// MI455X (gfx1250) fused GAT-intent kernel, round 3.
//
// Roofline: ~20 GFLOP encoder GEMM vs 174 MB neighbor_obs (~7.5us @23.3TB/s)
// -> bf16 WMMA with full fusion (no intent round-trip to HBM).
// Structure: pre-swizzled B-fragments (1 coalesced 32B/lane load each), each
// wave encodes one batch element's 32 neighbors as two M-tiles sharing every
// B-fragment (2 WMMAs per weight load, dual accumulation chains).
// Round-3 changes: (a) exp-based exact tanh (v_exp+v_rcp instead of libm
// tanhf chain) to shrink the VALU tail; (b) attention logits computed from
// in-register D-layout tanh values via shfl-reduction instead of strided
// LDS re-reads, dropping one barrier phase.
// Algebraic fold (softmax sums to 1):
//   out = ctx @ (Wv@Wo) + (bv@Wo + bo),  ctx = sum_n alpha_n * nbr_intent_n
// ---------------------------------------------------------------------------

typedef __attribute__((ext_vector_type(16))) __bf16 v16bf;
typedef __attribute__((ext_vector_type(8)))  __bf16 v8bf;
typedef __attribute__((ext_vector_type(8)))  float  v8f;

constexpr int OBSD = 83;    // obs dim (padded to 96 = 3 K-steps of 32)
constexpr int HIDD = 128;   // hidden dim (8 N-tiles; 4 K-steps for layer 2)
constexpr int INTD = 64;    // intent dim (4 N-tiles)
constexpr int NNB  = 32;    // neighbors

// A-matrix (16x32 bf16) per-lane element -> K mapping (contiguous, K = lh*16 + e, matches B fragments)
__device__ __forceinline__ int a_klocal(int e, int lh) {
    return lh * 16 + e;
}

// tanh(x) = 1 - 2/(exp(2x)+1): exact identity, saturates to +-1, maps to
// v_exp_f32 + v_rcp_f32 + 3 VALU instead of the libm tanhf chain.
__device__ __forceinline__ float fast_tanh(float x) {
    float e = __expf(2.0f * x);
    return 1.0f - __fdividef(2.0f, e + 1.0f);
}

// ---------------------------------------------------------------------------
// Single-tile encoder (used by the small self pass).
// ---------------------------------------------------------------------------
__device__ __forceinline__ void encode16(
    const float* __restrict__ rowbase,
    const v16bf* __restrict__ w1f, const v16bf* __restrict__ w2f,
    const float* __restrict__ b1, const float* __restrict__ b2,
    __bf16* hid, int lane, float tout[4][8])
{
    const int m  = lane & 15;
    const int lh = lane >> 4;
    const float* rowp = rowbase + (size_t)m * OBSD;

    v8f acc[8] = {};
#pragma unroll
    for (int kt = 0; kt < 3; ++kt) {
        v16bf a;
#pragma unroll
        for (int e = 0; e < 16; ++e) {
            int K = kt * 32 + a_klocal(e, lh);
            a[e] = (__bf16)((K < OBSD) ? rowp[K] : 0.0f);
        }
#pragma unroll
        for (int nt = 0; nt < 8; ++nt) {
            v16bf bf = w1f[(kt * 8 + nt) * 32 + lane];
            acc[nt] = __builtin_amdgcn_wmma_f32_16x16x32_bf16(
                false, a, false, bf, (short)0, acc[nt], false, false);
        }
    }
#pragma unroll
    for (int nt = 0; nt < 8; ++nt) {
        float bias = b1[nt * 16 + m];
        v8bf pk;
#pragma unroll
        for (int r = 0; r < 8; ++r)
            pk[r] = (__bf16)fmaxf(acc[nt][r] + bias, 0.0f);
        *(v8bf*)&hid[(nt * 16 + m) * 16 + lh * 8] = pk;
    }
    asm volatile("s_wait_dscnt 0" ::: "memory");

    v8f acc2[4] = {};
#pragma unroll
    for (int kt = 0; kt < 4; ++kt) {
        v16bf a2;
#pragma unroll
        for (int e = 0; e < 16; ++e)
            a2[e] = hid[(kt * 32 + a_klocal(e, lh)) * 16 + m];
#pragma unroll
        for (int nt = 0; nt < 4; ++nt) {
            v16bf bf = w2f[(kt * 4 + nt) * 32 + lane];
            acc2[nt] = __builtin_amdgcn_wmma_f32_16x16x32_bf16(
                false, a2, false, bf, (short)0, acc2[nt], false, false);
        }
    }
#pragma unroll
    for (int nt = 0; nt < 4; ++nt) {
        float bias = b2[nt * 16 + m];
#pragma unroll
        for (int r = 0; r < 8; ++r)
            tout[nt][r] = fast_tanh(acc2[nt][r] + bias);
    }
}

// ---------------------------------------------------------------------------
// Two-tile encoder: one wave encodes 32 rows; each B-fragment load feeds two
// WMMAs (halves weight bandwidth per row, dual accumulation chains).
// hid = this wave's [2][128][16] bf16 LDS slice (8 KB).
// ---------------------------------------------------------------------------
__device__ __forceinline__ void encode32(
    const float* __restrict__ row0, const float* __restrict__ row1,
    const v16bf* __restrict__ w1f, const v16bf* __restrict__ w2f,
    const float* __restrict__ b1, const float* __restrict__ b2,
    __bf16* hid, int lane, float tout[2][4][8])
{
    const int m  = lane & 15;
    const int lh = lane >> 4;
    const float* rp0 = row0 + (size_t)m * OBSD;
    const float* rp1 = row1 + (size_t)m * OBSD;

    // ---- layer 1: two 16x96 A-tiles against shared 96x128 B ----
    v8f acc0[8] = {}, acc1[8] = {};
#pragma unroll
    for (int kt = 0; kt < 3; ++kt) {
        v16bf a0, a1;
#pragma unroll
        for (int e = 0; e < 16; ++e) {
            int K = kt * 32 + a_klocal(e, lh);
            bool ok = (K < OBSD);
            a0[e] = (__bf16)(ok ? rp0[K] : 0.0f);
            a1[e] = (__bf16)(ok ? rp1[K] : 0.0f);
        }
#pragma unroll
        for (int nt = 0; nt < 8; ++nt) {
            v16bf bf = w1f[(kt * 8 + nt) * 32 + lane];   // loaded once
            acc0[nt] = __builtin_amdgcn_wmma_f32_16x16x32_bf16(
                false, a0, false, bf, (short)0, acc0[nt], false, false);
            acc1[nt] = __builtin_amdgcn_wmma_f32_16x16x32_bf16(
                false, a1, false, bf, (short)0, acc1[nt], false, false);
        }
    }

    // ---- bias + ReLU, pack D-layout rows into column-major hid slices ----
#pragma unroll
    for (int nt = 0; nt < 8; ++nt) {
        float bias = b1[nt * 16 + m];
        v8bf p0, p1;
#pragma unroll
        for (int r = 0; r < 8; ++r) {
            p0[r] = (__bf16)fmaxf(acc0[nt][r] + bias, 0.0f);
            p1[r] = (__bf16)fmaxf(acc1[nt][r] + bias, 0.0f);
        }
        *(v8bf*)&hid[(nt * 16 + m) * 16 + lh * 8]        = p0;
        *(v8bf*)&hid[2048 + (nt * 16 + m) * 16 + lh * 8] = p1;
    }
    asm volatile("s_wait_dscnt 0" ::: "memory");   // wave-private LDS handoff

    // ---- layer 2: two 16x128 A-tiles against shared 128x64 B ----
    v8f acc2a[4] = {}, acc2b[4] = {};
#pragma unroll
    for (int kt = 0; kt < 4; ++kt) {
        v16bf a0, a1;
#pragma unroll
        for (int e = 0; e < 16; ++e) {
            int K = kt * 32 + a_klocal(e, lh);
            a0[e] = hid[K * 16 + m];
            a1[e] = hid[2048 + K * 16 + m];
        }
#pragma unroll
        for (int nt = 0; nt < 4; ++nt) {
            v16bf bf = w2f[(kt * 4 + nt) * 32 + lane];   // loaded once
            acc2a[nt] = __builtin_amdgcn_wmma_f32_16x16x32_bf16(
                false, a0, false, bf, (short)0, acc2a[nt], false, false);
            acc2b[nt] = __builtin_amdgcn_wmma_f32_16x16x32_bf16(
                false, a1, false, bf, (short)0, acc2b[nt], false, false);
        }
    }
#pragma unroll
    for (int nt = 0; nt < 4; ++nt) {
        float bias = b2[nt * 16 + m];
#pragma unroll
        for (int r = 0; r < 8; ++r) {
            tout[0][nt][r] = fast_tanh(acc2a[nt][r] + bias);
            tout[1][nt][r] = fast_tanh(acc2b[nt][r] + bias);
        }
    }
}

// ---------------------------------------------------------------------------
// Prep: swizzle W1/W2 into WMMA B-fragment order (bf16), fold Wv@Wo.
// B-fragment order used here (contiguous k, same as a_klocal): lane -> N = lane&15, K = (lane>>4)*16 + e.
// ---------------------------------------------------------------------------
__global__ void __launch_bounds__(256) k_prep(
    const float* __restrict__ W1, const float* __restrict__ W2,
    const float* __restrict__ Wv, const float* __restrict__ bv,
    const float* __restrict__ Wo, const float* __restrict__ bo,
    __bf16* __restrict__ w1f, __bf16* __restrict__ w2f,
    float* __restrict__ Wvo, float* __restrict__ bvo)
{
    int tid = threadIdx.x;
    for (int idx = tid; idx < 3 * 8 * 32 * 16; idx += 256) {
        int e = idx & 15, lane = (idx >> 4) & 31, nt = (idx >> 9) & 7, kt = idx >> 12;
        int n = nt * 16 + (lane & 15);
        int K = kt * 32 + (lane >> 4) * 16 + e;
        w1f[idx] = (__bf16)((K < OBSD) ? W1[K * HIDD + n] : 0.0f);
    }
    for (int idx = tid; idx < 4 * 4 * 32 * 16; idx += 256) {
        int e = idx & 15, lane = (idx >> 4) & 31, nt = (idx >> 9) & 3, kt = idx >> 11;
        int n = nt * 16 + (lane & 15);
        int K = kt * 32 + (lane >> 4) * 16 + e;
        w2f[idx] = (__bf16)W2[K * INTD + n];
    }
    for (int idx = tid; idx < 64 * 16; idx += 256) {
        int k = idx >> 4, j = idx & 15;
        float s = 0.0f;
        for (int mm = 0; mm < 64; ++mm) s += Wv[k * 64 + mm] * Wo[mm * 16 + j];
        Wvo[idx] = s;
    }
    if (tid < 16) {
        float s = bo[tid];
        for (int mm = 0; mm < 64; ++mm) s += bv[mm] * Wo[mm * 16 + tid];
        bvo[tid] = s;
    }
}

// ---------------------------------------------------------------------------
// Self pass (~3% of FLOPs): encode 16 self rows/wave, reduce against
// Wa[0:64] (the only use of self intent): sd[b] = dot(intent, Wa_s) + ba.
// ---------------------------------------------------------------------------
__global__ void __launch_bounds__(256) k_self(
    const float* __restrict__ self_obs,
    const v16bf* __restrict__ w1f, const v16bf* __restrict__ w2f,
    const float* __restrict__ b1, const float* __restrict__ b2,
    const float* __restrict__ Wa, const float* __restrict__ ba,
    float* __restrict__ sd)
{
    __shared__ __bf16 hid[8][HIDD * 16];
    int tid = threadIdx.x, lane = tid & 31, w = tid >> 5;
    int m = lane & 15, lh = lane >> 4;
    int tile = blockIdx.x * 8 + w;

    float t[4][8];
    encode16(self_obs + (size_t)tile * 16 * OBSD, w1f, w2f, b1, b2,
             hid[w], lane, t);

    float was[4];
#pragma unroll
    for (int nt = 0; nt < 4; ++nt) was[nt] = Wa[nt * 16 + m];
    float ba0 = ba[0];
#pragma unroll
    for (int r = 0; r < 8; ++r) {
        float p = 0.0f;
#pragma unroll
        for (int nt = 0; nt < 4; ++nt) p += t[nt][r] * was[nt];
#pragma unroll
        for (int off = 1; off < 16; off <<= 1) p += __shfl_xor(p, off, 32);
        if (m == 0) sd[tile * 16 + lh * 8 + r] = p + ba0;  // row M = lh*8+r
    }
}

// ---------------------------------------------------------------------------
// Main fused kernel: 128 threads = 4 waves, 4 batch elements per block; each
// wave encodes one batch element's 32 neighbors (two shared-B M-tiles), then
// in-register attention logits, in-wave softmax, context, folded projection.
// ---------------------------------------------------------------------------
__global__ void __launch_bounds__(128) k_main(
    const float* __restrict__ nbr, const float* __restrict__ ew,
    const v16bf* __restrict__ w1f, const v16bf* __restrict__ w2f,
    const float* __restrict__ b1, const float* __restrict__ b2,
    const float* __restrict__ Wa,
    const float* __restrict__ Wvo, const float* __restrict__ bvo,
    const float* __restrict__ sd, float* __restrict__ out)
{
    __shared__ __bf16 hid[4][2 * HIDD * 16];       // 32 KB (per-wave slices)
    __shared__ __bf16 intent[4][INTD][NNB];        // 16 KB, [b][k][n]
    __shared__ float  dlog[4][NNB];
    __shared__ float  alpha[4][NNB];
    __shared__ float  ctx[4][INTD];

    int tid = threadIdx.x, lane = tid & 31, w = tid >> 5;
    int m = lane & 15, lh = lane >> 4;
    int bg = blockIdx.x;

    const float* base = nbr + (size_t)(bg * 4 + w) * NNB * OBSD;
    float t[2][4][8];
    encode32(base, base + 16 * OBSD, w1f, w2f, b1, b2, hid[w], lane, t);

    // intents column-major [k][n]: D-layout rows pack into b128 stores
#pragma unroll
    for (int tt = 0; tt < 2; ++tt) {
#pragma unroll
        for (int nt = 0; nt < 4; ++nt) {
            v8bf pk;
#pragma unroll
            for (int r = 0; r < 8; ++r) pk[r] = (__bf16)t[tt][nt][r];
            *(v8bf*)&intent[w][nt * 16 + m][tt * 16 + lh * 8] = pk;
        }
    }

    // attention logits straight from D-layout registers (as in k_self):
    // neighbor n = tt*16 + lh*8 + r, dot over k = nt*16 + m via shfl tree
    {
        float wan[4];
#pragma unroll
        for (int nt = 0; nt < 4; ++nt) wan[nt] = Wa[64 + nt * 16 + m];
#pragma unroll
        for (int tt = 0; tt < 2; ++tt) {
#pragma unroll
            for (int r = 0; r < 8; ++r) {
                float p = 0.0f;
#pragma unroll
                for (int nt = 0; nt < 4; ++nt) p += t[tt][nt][r] * wan[nt];
#pragma unroll
                for (int off = 1; off < 16; off <<= 1) p += __shfl_xor(p, off, 32);
                if (m == 0) dlog[w][tt * 16 + lh * 8 + r] = p;
            }
        }
    }
    asm volatile("s_wait_dscnt 0" ::: "memory");   // wave-private dlog handoff

    // in-wave softmax over 32 neighbors (wave w owns batch-local w)
    {
        int n = lane;
        float sdb = sd[bg * 4 + w];                // uniform per wave
        float e = dlog[w][n] + sdb;
        e = (e > 0.0f) ? e : 0.2f * e;             // leaky_relu(0.2)
        e *= ew[(size_t)(bg * 4 + w) * NNB + n];
        float mx = e;
#pragma unroll
        for (int off = 1; off < 32; off <<= 1) mx = fmaxf(mx, __shfl_xor(mx, off, 32));
        float ex = __expf(e - mx);
        float sm = ex;
#pragma unroll
        for (int off = 1; off < 32; off <<= 1) sm += __shfl_xor(sm, off, 32);
        alpha[w][n] = ex / sm;
    }
    __syncthreads();   // alpha + intent visible block-wide

    // ctx[b][k] = sum_n alpha * intent  (each thread: 2 k-columns)
    {
        int b = tid >> 5;
#pragma unroll
        for (int kk = 0; kk < 2; ++kk) {
            int k = (tid & 31) + kk * 32;
            float c = 0.0f;
#pragma unroll
            for (int n = 0; n < NNB; ++n) c += alpha[b][n] * (float)intent[b][k][n];
            ctx[b][k] = c;
        }
    }
    __syncthreads();

    // out = ctx @ Wvo + bvo  (folded Wv@Wo — ~0.15% of total FLOPs)
    if (tid < 64) {
        int b = tid >> 4, j = tid & 15;
        float o = bvo[j];
#pragma unroll
        for (int k = 0; k < INTD; ++k) o += ctx[b][k] * Wvo[k * 16 + j];
        out[(size_t)(bg * 4 + b) * 16 + j] = o;
    }
}

// ---------------------------------------------------------------------------
extern "C" void kernel_launch(void* const* d_in, const int* in_sizes, int n_in,
                              void* d_out, int out_size, void* d_ws, size_t ws_size,
                              hipStream_t stream)
{
    const float* self_obs = (const float*)d_in[0];
    const float* nbr      = (const float*)d_in[1];
    const float* ew       = (const float*)d_in[2];
    const float* W1 = (const float*)d_in[3];
    const float* b1 = (const float*)d_in[4];
    const float* W2 = (const float*)d_in[5];
    const float* b2 = (const float*)d_in[6];
    const float* Wa = (const float*)d_in[7];
    const float* ba = (const float*)d_in[8];
    const float* Wv = (const float*)d_in[9];
    const float* bv = (const float*)d_in[10];
    const float* Wo = (const float*)d_in[11];
    const float* bo = (const float*)d_in[12];
    float* out = (float*)d_out;

    const int B = in_sizes[0] / OBSD;   // 16384

    char* ws = (char*)d_ws;
    __bf16* w1f = (__bf16*)(ws);             // 24576 B: layer-1 B-fragments
    __bf16* w2f = (__bf16*)(ws + 24576);     // 16384 B: layer-2 B-fragments
    float*  Wvo = (float*)(ws + 40960);      //  4096 B: folded Wv@Wo
    float*  bvo = (float*)(ws + 45056);      //    64 B
    float*  sdv = (float*)(ws + 45120);      //  B*4 B: self-side logit term

    k_prep<<<1, 256, 0, stream>>>(W1, W2, Wv, bv, Wo, bo, w1f, w2f, Wvo, bvo);
    k_self<<<B / 128, 256, 0, stream>>>(self_obs, (const v16bf*)w1f,
                                        (const v16bf*)w2f, b1, b2, Wa, ba, sdv);
    k_main<<<B / 4, 128, 0, stream>>>(nbr, ew, (const v16bf*)w1f,
                                      (const v16bf*)w2f, b1, b2, Wa,
                                      Wvo, bvo, sdv, out);
}
